// ModuleCorrelation_5669356831807
// MI455X (gfx1250) — matrix operand order repaired and checked by reading, not compiled, not run on hardware
//
#include <hip/hip_runtime.h>

#define PADV 4
#define DV   9           // displacements per axis
#define BB   8
#define CC   128
#define HH   96
#define WW   160

typedef __bf16 bf16;
typedef __attribute__((ext_vector_type(16))) __bf16 v16bf;
typedef __attribute__((ext_vector_type(8)))  __bf16 v8bf;
typedef __attribute__((ext_vector_type(8)))  float  v8f;
// 16-byte-aligned view of a 32-byte B fragment (DS b128 needs only 16B alignment)
typedef v16bf __attribute__((aligned(16))) v16bf_a16;

#define CP        136            // channel pitch in halves: 272B = 17*16B -> aligned + conflict-free
#define TWO_ROWS  16             // y0-4 .. y0+11
#define TWO_COLS  32             // x0-4 .. x0+27 (cols >= 24 stay zero)
#define ONE_ROWS  8
#define ONE_COLS  16

#define TWO_S_BYTES (TWO_ROWS*TWO_COLS*CP*2)
#define ONE_S_BYTES (ONE_ROWS*ONE_COLS*CP*2)
#define G_BYTES     (8*16*32*4)
#define SMEM_BYTES  (TWO_S_BYTES + ONE_S_BYTES + G_BYTES)

__global__ __launch_bounds__(256) void corr81_wmma_kernel(const float* __restrict__ one,
                                                          const float* __restrict__ two,
                                                          float* __restrict__ out) {
  extern __shared__ char smem[];
  bf16*  two_s = (bf16*)smem;
  bf16*  one_s = (bf16*)(smem + TWO_S_BYTES);
  float* G     = (float*)(smem + TWO_S_BYTES + ONE_S_BYTES);

  const int tid = threadIdx.x;
  const int b   = blockIdx.z;
  const int y0  = blockIdx.y * ONE_ROWS;
  const int x0  = blockIdx.x * ONE_COLS;
  const int HW  = HH * WW;

  // ---- stage `one` tile -> LDS bf16 [yl][x][c], pairs of channels per b32 store ----
  for (int it = 0; it < 32; ++it) {
    int idx = it * 256 + tid;
    int x   = idx & 15;
    int cp  = (idx >> 4) & 63;
    int yl  = idx >> 10;
    const float* g = one + (((size_t)b * CC + 2 * cp) * HH + (y0 + yl)) * WW + (x0 + x);
    float f0 = g[0];
    float f1 = g[HW];
    union { unsigned int u; bf16 h[2]; } pk;
    pk.h[0] = (bf16)f0; pk.h[1] = (bf16)f1;
    *(unsigned int*)(one_s + (yl * ONE_COLS + x) * CP + 2 * cp) = pk.u;
  }

  // ---- stage `two` tile -> LDS bf16 [r][j][c], zero-filled at image / tile borders ----
  for (int it = 0; it < 128; ++it) {
    int idx = it * 256 + tid;
    int j   = idx & 31;
    int cp  = (idx >> 5) & 63;
    int r   = idx >> 11;
    int yg  = y0 - PADV + r;
    int xg  = x0 - PADV + j;
    float f0 = 0.0f, f1 = 0.0f;
    if (j < 24 && yg >= 0 && yg < HH && xg >= 0 && xg < WW) {
      const float* g = two + (((size_t)b * CC + 2 * cp) * HH + yg) * WW + xg;
      f0 = g[0];
      f1 = g[HW];
    }
    union { unsigned int u; bf16 h[2]; } pk;
    pk.h[0] = (bf16)f0; pk.h[1] = (bf16)f1;
    *(unsigned int*)(two_s + (r * TWO_COLS + j) * CP + 2 * cp) = pk.u;
  }

  __syncthreads();

  const int w    = tid >> 5;   // wave id == local y row
  const int l    = tid & 31;
  const int xl16 = l & 15;
  const int half = l >> 4;

  // ---- preload A fragments (contiguous k = 16*half + s per chunk, same as B; 4 K-chunks of 32) ----
  v16bf a_frag[4];
  const bf16* arow = one_s + (w * ONE_COLS + xl16) * CP;
  #pragma unroll
  for (int k = 0; k < 4; ++k) {
    v8bf lo = *(const v8bf*)(arow + k * 32 + half * 16);       // K = 32k + 16*half + 0..7
    v8bf hi = *(const v8bf*)(arow + k * 32 + half * 16 + 8);   // K = 32k + 16*half + 8..15
    a_frag[k] = __builtin_shufflevector(lo, hi, 0,1,2,3,4,5,6,7,8,9,10,11,12,13,14,15);
  }

  float* Gw = G + w * 16 * 32;   // wave-private 16x32 f32 scratch
  const float inv = 1.0f / (float)CC;

  for (int p = 0; p < DV; ++p) {
    const bf16* brow0 = two_s + ((w + p) * TWO_COLS + xl16) * CP;  // N-tile0: col j = xl16
    const bf16* brow1 = brow0 + 16 * CP;                            // N-tile1: col j = 16+xl16

    // ---- issue ALL 16 ds_load_b128 for this p, pinned in 8 distinct register
    //      tuples so a single s_wait_dscnt covers the whole WMMA chain ----
    int off1 = half * 16;           // B layout: lanes 0-15 K=0..15, lanes 16-31 K=16..31
    v16bf b0 = *(const v16bf_a16*)(brow0 + off1);
    v16bf b1 = *(const v16bf_a16*)(brow1 + off1);
    v16bf b2 = *(const v16bf_a16*)(brow0 + 32 + off1);
    v16bf b3 = *(const v16bf_a16*)(brow1 + 32 + off1);
    v16bf b4 = *(const v16bf_a16*)(brow0 + 64 + off1);
    v16bf b5 = *(const v16bf_a16*)(brow1 + 64 + off1);
    v16bf b6 = *(const v16bf_a16*)(brow0 + 96 + off1);
    v16bf b7 = *(const v16bf_a16*)(brow1 + 96 + off1);
    // scheduling fence: force all fragments materialized (defeats load re-sinking)
    asm volatile("" : "+v"(b0), "+v"(b1), "+v"(b2), "+v"(b3),
                      "+v"(b4), "+v"(b5), "+v"(b6), "+v"(b7));

    v8f acc0 = {};
    v8f acc1 = {};
    acc0 = __builtin_amdgcn_wmma_f32_16x16x32_bf16(false, a_frag[0], false, b0, (short)0, acc0, false, false);
    acc1 = __builtin_amdgcn_wmma_f32_16x16x32_bf16(false, a_frag[0], false, b1, (short)0, acc1, false, false);
    acc0 = __builtin_amdgcn_wmma_f32_16x16x32_bf16(false, a_frag[1], false, b2, (short)0, acc0, false, false);
    acc1 = __builtin_amdgcn_wmma_f32_16x16x32_bf16(false, a_frag[1], false, b3, (short)0, acc1, false, false);
    acc0 = __builtin_amdgcn_wmma_f32_16x16x32_bf16(false, a_frag[2], false, b4, (short)0, acc0, false, false);
    acc1 = __builtin_amdgcn_wmma_f32_16x16x32_bf16(false, a_frag[2], false, b5, (short)0, acc1, false, false);
    acc0 = __builtin_amdgcn_wmma_f32_16x16x32_bf16(false, a_frag[3], false, b6, (short)0, acc0, false, false);
    acc1 = __builtin_amdgcn_wmma_f32_16x16x32_bf16(false, a_frag[3], false, b7, (short)0, acc1, false, false);

    // ---- spill D (M = v + 8*half, N = lane%16) to wave-private LDS scratch ----
    #pragma unroll
    for (int v = 0; v < 8; ++v) {
      Gw[(v + 8 * half) * 32 + xl16]      = acc0[v];
      Gw[(v + 8 * half) * 32 + 16 + xl16] = acc1[v];
    }

    // ---- gather diagonal band out[x,o] = G[x, x+o] / C and store (coalesced in x) ----
    #pragma unroll
    for (int i = 0; i < 5; ++i) {
      int idx = i * 32 + l;
      if (idx < 144) {
        int o  = idx >> 4;
        int xl = idx & 15;
        float gval = Gw[xl * 32 + xl + o];   // lane stride 33 dwords: conflict-free
        float* dst = out + (((size_t)b * 81 + p * DV + o) * HH + (y0 + w)) * WW + (x0 + xl);
        __builtin_nontemporal_store(gval * inv, dst);   // write-once output: NT hint
      }
    }
  }
}

extern "C" void kernel_launch(void* const* d_in, const int* in_sizes, int n_in,
                              void* d_out, int out_size, void* d_ws, size_t ws_size,
                              hipStream_t stream) {
  const float* one = (const float*)d_in[0];
  const float* two = (const float*)d_in[1];
  float* out = (float*)d_out;
  dim3 grid(WW / ONE_COLS, HH / ONE_ROWS, BB);   // (10, 12, 8)
  corr81_wmma_kernel<<<grid, 256, SMEM_BYTES, stream>>>(one, two, out);
}
